// DOMINO_2233382994514
// MI455X (gfx1250) — matrix operand order repaired and checked by reading, not compiled, not run on hardware
//
#include <hip/hip_runtime.h>
#include <cstdint>

typedef __attribute__((ext_vector_type(16))) _Float16 v16h;
typedef __attribute__((ext_vector_type(8)))  float    v8f;

#define NC 16                         // classes (fixed: matches WMMA 16x16 tile)
#define NB 8                          // batch
#define NHW (512*512)                 // pixels per image
#define BLOCKS_PER_BATCH 64
#define WPB 8                         // waves per 256-thread block (wave32)
#define WAVES_PER_BATCH (BLOCKS_PER_BATCH*WPB)   // 512
#define CHUNKS (NHW/32)               // 8192 32-pixel chunks per image
#define SMOOTH_NR 1e-5f
#define LOG2E 1.44269504088896340736f
#define LN2   0.69314718055994530942f

// ---------------------------------------------------------------- zero scratch
__global__ __launch_bounds__(256) void domino_zero(uint32_t* ws, int n) {
  int i = blockIdx.x * blockDim.x + threadIdx.x;
  if (i < n) ws[i] = 0u;
}

// ---------------------------------------------------------------- main pass
// One block = one batch image slice. Per 32-pixel chunk per wave:
//   softmax in f32 registers (non-temporal streaming loads) -> LDS staging ->
//   CE term read back via one dynamic ds_load (no 16-way register select) ->
//   Q_tile += OneHot(16x32,f16) x Probs(32x16,f16) via v_wmma_f32_16x16x32_f16
__global__ __launch_bounds__(256) void domino_main(
    const float* __restrict__ logits,   // [B][C][H*W]
    const int*   __restrict__ targets,  // [B][H*W]
    float* __restrict__ Qg,             // [B][16][16]  Q_b[t][c]
    int*   __restrict__ cntg,           // [B][16]      pixel counts per class
    float* __restrict__ ceg)            // [1]          sum of -log p_t
{
  __shared__ float lds_p[WPB][NC][33];  // probs staging, 33-stride => conflict-free
  __shared__ int   lds_t[WPB][32];      // targets of the wave's 32 pixels
  __shared__ float lds_Q[256];          // block-level 16x16 Q reduction
  __shared__ int   lds_cnt[NC];
  __shared__ float lds_ce;

  const int tid  = threadIdx.x;
  const int wave = tid >> 5;
  const int lane = tid & 31;
  const int half = lane >> 4;           // which 16-lane half
  const int n    = lane & 15;           // row (A) / column (B/D) index

  lds_Q[tid] = 0.0f;
  if (tid < NC) lds_cnt[tid] = 0;
  if (tid == 0) lds_ce = 0.0f;
  __syncthreads();

  const int b      = blockIdx.x / BLOCKS_PER_BATCH;
  const int blkInB = blockIdx.x % BLOCKS_PER_BATCH;
  const int wInB   = blkInB * WPB + wave;

  const float* bbase = logits  + (size_t)b * NC * NHW;
  const int*   tbase = targets + (size_t)b * NHW;

  v8f   acc = {};          // D/C tile: Q[t][c] accumulated in f32
  float ce_local = 0.0f;

  for (int chunk = wInB; chunk < CHUNKS; chunk += WAVES_PER_BATCH) {
    const int hw = (chunk << 5) + lane;         // this lane's pixel
    const int t  = __builtin_nontemporal_load(&tbase[hw]);

    // -------- softmax over 16 channels (coalesced; NT: 128MB read-once stream)
    float x[NC];
    float m = -3.402823466e+38f;
#pragma unroll
    for (int c = 0; c < NC; ++c) {
      x[c] = __builtin_nontemporal_load(&bbase[(size_t)c * NHW + hw]);
      m = fmaxf(m, x[c]);
    }
    const float mL = m * LOG2E;                  // fold max-sub into one FMA
    float s = 0.0f;
    float p[NC];
#pragma unroll
    for (int c = 0; c < NC; ++c) {
      p[c] = __builtin_amdgcn_exp2f(__builtin_fmaf(x[c], LOG2E, -mL));
      s += p[c];
    }
    const float inv = 1.0f / s;

    // -------- stage normalized probs (f32) + targets in per-wave LDS tiles
#pragma unroll
    for (int c = 0; c < NC; ++c) lds_p[wave][c][lane] = p[c] * inv;
    lds_t[wave][lane] = t;
    atomicAdd(&lds_cnt[t], 1);                   // exact onehot counts

    __builtin_amdgcn_wave_barrier();             // same-wave LDS RAW: order sched
    asm volatile("" ::: "memory");

    // CE: -log p_t via one dynamic LDS read (replaces 16x cmp+cndmask chain)
    const float pt = lds_p[wave][t][lane];
    ce_local -= LN2 * __builtin_amdgcn_logf(pt); // v_log_f32 is log2

    // -------- build WMMA operands in the documented gfx1250 VGPR layouts.
    // A (16x32 f16, onehot): lane holds row M=n; K per VGPR v:
    //   lanes 0-15: K = {2v, 2v+1} (v<4), {2v+8, 2v+9} (v>=4);  lanes 16-31: +8
    // B (32x16 f16, probs): lane holds col N=n; VGPR v holds same K as A
    v16h A, Bm;
#pragma unroll
    for (int v = 0; v < 8; ++v) {
      const int ka  = 2*v + ((v >= 4) ? 8 : 0) + 8*half;
      const int ta  = lds_t[wave][ka];
      const int tb_ = lds_t[wave][ka + 1];
      A[2*v]   = (ta  == n) ? (_Float16)1.0f : (_Float16)0.0f;
      A[2*v+1] = (tb_ == n) ? (_Float16)1.0f : (_Float16)0.0f;
      const int kb = 2*v + ((v >= 4) ? 8 : 0) + 8*half;
      Bm[2*v]   = (_Float16)lds_p[wave][n][kb];
      Bm[2*v+1] = (_Float16)lds_p[wave][n][kb + 1];
    }

    // Q_tile += A x B   (EXEC all-ones: uniform trip count, full block)
    acc = __builtin_amdgcn_wmma_f32_16x16x32_f16(false, A, false, Bm,
                                                 (short)0, acc, false, false);

    // one speculative prefetch for the next chunk (global_prefetch_b8)
    if (chunk + WAVES_PER_BATCH < CHUNKS)
      __builtin_prefetch(&bbase[hw + WAVES_PER_BATCH*32], 0, 0);
  }

  // -------- fold wave Q tile into block tile (D layout: VGPR j -> M=j+8*half, N=n)
#pragma unroll
  for (int j = 0; j < 8; ++j)
    atomicAdd(&lds_Q[(j + 8*half)*16 + n], acc[j]);

  // CE: wave32 butterfly reduction, then one LDS atomic per wave
#pragma unroll
  for (int off = 16; off > 0; off >>= 1)
    ce_local += __shfl_xor(ce_local, off, 32);
  if (lane == 0) atomicAdd(&lds_ce, ce_local);

  __syncthreads();

  // block -> global (per-batch) accumulators
  atomicAdd(&Qg[b*256 + tid], lds_Q[tid]);
  if (tid < NC) atomicAdd(&cntg[b*NC + tid], lds_cnt[tid]);
  if (tid == 0) atomicAdd(ceg, lds_ce);
}

// ---------------------------------------------------------------- finalize
// CE + Dice + penalty from the 8x16x16 sufficient statistic. One wave.
__global__ __launch_bounds__(32) void domino_finalize(
    const float* __restrict__ Qg, const int* __restrict__ cntg,
    const float* __restrict__ ceg, const float* __restrict__ MP,
    float* __restrict__ out)
{
  const int lane = threadIdx.x;
  float dice_s = 0.0f, pen_s = 0.0f;
  for (int idx = lane; idx < NB*NC; idx += 32) {
    const int bb = idx >> 4, c = idx & 15;
    const float* Qb = Qg + bb*256;
    float colsum = 0.0f, pen = 0.0f;
#pragma unroll
    for (int t = 0; t < NC; ++t) {
      const float q = Qb[t*16 + c];
      colsum += q;                       // sum_p probs[b,:,c]
      pen    += MP[t*16 + c] * q;        // DOMINO contraction
    }
    const float inter = Qb[c*16 + c];    // sum probs*onehot
    const float cnt   = (float)cntg[bb*NC + c];
    dice_s += 1.0f - (2.0f*inter + SMOOTH_NR) / (colsum + cnt + SMOOTH_NR);
    pen_s  += pen;
  }
#pragma unroll
  for (int off = 16; off > 0; off >>= 1) {
    dice_s += __shfl_xor(dice_s, off, 32);
    pen_s  += __shfl_xor(pen_s,  off, 32);
  }
  if (lane == 0) {
    const float ce = ceg[0] / (float)((size_t)NB * NHW);
    out[0] = ce + dice_s / (float)(NB*NC) + pen_s / (float)NB;   // beta=1, /total_batch
  }
}

// ---------------------------------------------------------------- launch
extern "C" void kernel_launch(void* const* d_in, const int* in_sizes, int n_in,
                              void* d_out, int out_size, void* d_ws, size_t ws_size,
                              hipStream_t stream) {
  (void)in_sizes; (void)n_in; (void)out_size; (void)ws_size;
  const float* logits  = (const float*)d_in[0];
  const int*   targets = (const int*)d_in[1];
  const float* MP      = (const float*)d_in[2];

  float* Qg  = (float*)d_ws;                                          // 8*256 f32
  int*   cnt = (int*)  ((char*)d_ws + NB*256*sizeof(float));          // 8*16 i32
  float* ce  = (float*)((char*)d_ws + NB*256*sizeof(float) + NB*NC*sizeof(int));

  const int zero_dwords = NB*256 + NB*NC + 1;
  domino_zero<<<(zero_dwords + 255)/256, 256, 0, stream>>>((uint32_t*)d_ws, zero_dwords);
  domino_main<<<NB*BLOCKS_PER_BATCH, 256, 0, stream>>>(logits, targets, Qg, cnt, ce);
  domino_finalize<<<1, 32, 0, stream>>>(Qg, cnt, ce, MP, (float*)d_out);
}
